// SFullAttention_52415780880837
// MI455X (gfx1250) — matrix operand order repaired and checked by reading, not compiled, not run on hardware
//
#include <hip/hip_runtime.h>

typedef _Float16 v16h __attribute__((ext_vector_type(16)));
typedef _Float16 v8h  __attribute__((ext_vector_type(8)));
typedef float    v8f  __attribute__((ext_vector_type(8)));

#define WMMA_F16(A, B, C) \
  __builtin_amdgcn_wmma_f32_16x16x32_f16(false, (A), false, (B), (short)0, (C), false, false)

// Problem constants (B,N,L,H,E) = (4,256,64,8,64), M=N, D=E
constexpr int Bb = 4, Nn = 256, Ll = 64, Hh = 8, Ee = 64;
constexpr int NSTR = Ll * Hh * Ee;   // stride between spatial rows = 32768 floats
constexpr int TKV  = 128;            // keys per LDS tile (2 tiles cover M=256)

// ---- Single-instruction DPP reductions: the DPP modifier rides on src0 of
// the max/add itself (v_max_num_f32_dpp vd, dpp(v), v). xor-pattern within
// 16-lane rows: quad_perm for xor1/xor2, row_xmask for xor4/xor8.
#define DPP_MAX_STEP(v, PAT)                                             \
  {                                                                      \
    float _d;                                                            \
    asm("v_max_num_f32_dpp %0, %1, %1 " PAT                              \
        " row_mask:0xf bank_mask:0xf"                                    \
        : "=v"(_d) : "v"(v));                                            \
    (v) = _d;                                                            \
  }
#define DPP_ADD_STEP(v, PAT)                                             \
  {                                                                      \
    float _d;                                                            \
    asm("v_add_f32_dpp %0, %1, %1 " PAT                                  \
        " row_mask:0xf bank_mask:0xf"                                    \
        : "=v"(_d) : "v"(v));                                            \
    (v) = _d;                                                            \
  }

__device__ __forceinline__ float rmax16(float v) {
  DPP_MAX_STEP(v, "quad_perm:[1,0,3,2]")
  DPP_MAX_STEP(v, "quad_perm:[2,3,0,1]")
  DPP_MAX_STEP(v, "row_xmask:4")
  DPP_MAX_STEP(v, "row_xmask:8")
  return v;
}
__device__ __forceinline__ float rsum16(float v) {
  DPP_ADD_STEP(v, "quad_perm:[1,0,3,2]")
  DPP_ADD_STEP(v, "quad_perm:[2,3,0,1]")
  DPP_ADD_STEP(v, "row_xmask:4")
  DPP_ADD_STEP(v, "row_xmask:8")
  return v;
}
// Raw v_max_num (no sNaN canonicalization spam from strict fmaxf).
__device__ __forceinline__ float fmax_raw(float a, float b) {
  float d;
  asm("v_max_num_f32 %0, %1, %2" : "=v"(d) : "v"(a), "v"(b));
  return d;
}

__launch_bounds__(256, 1)
__global__ void sfa_flash_kernel(const float* __restrict__ Q,
                                 const float* __restrict__ K,
                                 const float* __restrict__ V,
                                 float* __restrict__ O) {
  // LDS: K tile row-major f16 (B-frags for Q*K^T), V tile transposed f16
  // (B-frags for P*V), per-wave P staging (D-layout -> A-layout fixup).
  __shared__ __align__(32) _Float16 Klds[TKV * Ee];    // 16 KB
  __shared__ __align__(32) _Float16 Vt[Ee * TKV];      // 16 KB
  __shared__ __align__(32) _Float16 Pst[8][16 * 32];   //  8 KB

  const int bid = blockIdx.x;                 // (b,l,h) flattened: 2048 blocks
  const int h = bid % Hh;
  const int l = (bid / Hh) % Ll;
  const int b = bid / (Hh * Ll);
  const size_t base = (size_t)b * Nn * NSTR + (size_t)l * Hh * Ee + (size_t)h * Ee;

  const int tid  = threadIdx.x;
  const int wave = tid >> 5;
  const int lane = tid & 31;
  const int ml   = lane & 15;   // column index within 16-wide fragment
  const int g    = lane >> 4;   // lane-group (row/K half select)

  const float* qb = Q + base;
  const float* kb_g = K + base;
  const float* vb_g = V + base;
  float* ob = O + base;

  // ---- Load this wave's 32 query rows as f16 A-fragments.
  // Fold (1/sqrt(E)) * log2(e) into Q: softmax is base-invariant once the
  // scale absorbs log2(e), so the hot loop uses raw v_exp_f32 (exp2).
  const float scale = 0.125f * 1.44269504088896340736f;
  v16h qa[2][2];
#pragma unroll
  for (int t = 0; t < 2; ++t) {
    const float* qrow = qb + (size_t)(wave * 32 + t * 16 + ml) * NSTR;
#pragma unroll
    for (int ec = 0; ec < 2; ++ec) {
      const int e0 = ec * 32 + g * 16;
      float4 a0 = *(const float4*)(qrow + e0);
      float4 a1 = *(const float4*)(qrow + e0 + 4);
      float4 b0 = *(const float4*)(qrow + e0 + 8);
      float4 b1 = *(const float4*)(qrow + e0 + 12);
      v16h f;
      f[0]  = (_Float16)(a0.x * scale); f[1]  = (_Float16)(a0.y * scale);
      f[2]  = (_Float16)(a0.z * scale); f[3]  = (_Float16)(a0.w * scale);
      f[4]  = (_Float16)(a1.x * scale); f[5]  = (_Float16)(a1.y * scale);
      f[6]  = (_Float16)(a1.z * scale); f[7]  = (_Float16)(a1.w * scale);
      f[8]  = (_Float16)(b0.x * scale); f[9]  = (_Float16)(b0.y * scale);
      f[10] = (_Float16)(b0.z * scale); f[11] = (_Float16)(b0.w * scale);
      f[12] = (_Float16)(b1.x * scale); f[13] = (_Float16)(b1.y * scale);
      f[14] = (_Float16)(b1.z * scale); f[15] = (_Float16)(b1.w * scale);
      qa[t][ec] = f;
    }
  }

  // ---- Online-softmax state (C/D layout: lane holds 8 rows m = r + 8*g).
  float mrow[2][8], lrow[2][8];
  v8f oacc[2][4] = {};
#pragma unroll
  for (int t = 0; t < 2; ++t)
#pragma unroll
    for (int r = 0; r < 8; ++r) { mrow[t][r] = -1.0e30f; lrow[t][r] = 0.0f; }

  for (int kv = 0; kv < 2; ++kv) {
    // ---- Cooperative KV tile load: 128 keys, f32 -> f16, V transposed.
    {
      const int row  = tid >> 1;     // 0..127
      const int half = tid & 1;      // which 32 of the 64 channels
      const float* kr = kb_g + (size_t)(kv * TKV + row) * NSTR + half * 32;
      const float* vr = vb_g + (size_t)(kv * TKV + row) * NSTR + half * 32;
#pragma unroll
      for (int i = 0; i < 4; ++i) {
        float4 k0 = *(const float4*)(kr + i * 8);
        float4 k1 = *(const float4*)(kr + i * 8 + 4);
        v8h kh;
        kh[0] = (_Float16)k0.x; kh[1] = (_Float16)k0.y;
        kh[2] = (_Float16)k0.z; kh[3] = (_Float16)k0.w;
        kh[4] = (_Float16)k1.x; kh[5] = (_Float16)k1.y;
        kh[6] = (_Float16)k1.z; kh[7] = (_Float16)k1.w;
        *(v8h*)&Klds[row * Ee + half * 32 + i * 8] = kh;

        float4 v0 = *(const float4*)(vr + i * 8);
        float4 v1 = *(const float4*)(vr + i * 8 + 4);
        const int e0 = half * 32 + i * 8;
        Vt[(e0 + 0) * TKV + row] = (_Float16)v0.x;
        Vt[(e0 + 1) * TKV + row] = (_Float16)v0.y;
        Vt[(e0 + 2) * TKV + row] = (_Float16)v0.z;
        Vt[(e0 + 3) * TKV + row] = (_Float16)v0.w;
        Vt[(e0 + 4) * TKV + row] = (_Float16)v1.x;
        Vt[(e0 + 5) * TKV + row] = (_Float16)v1.y;
        Vt[(e0 + 6) * TKV + row] = (_Float16)v1.z;
        Vt[(e0 + 7) * TKV + row] = (_Float16)v1.w;
      }
    }
    __syncthreads();

    // ---- Process tile: 32 keys per inner step.
    for (int jj = 0; jj < TKV; jj += 32) {
      // B-frags for S = Q*K^T: column = key (jj + kc*16 + ml), K-dim = e
      // (contiguous 16 halves per lane from row-major K tile).
      v16h kfr[2][2];
#pragma unroll
      for (int kc = 0; kc < 2; ++kc)
#pragma unroll
        for (int ec = 0; ec < 2; ++ec)
          kfr[kc][ec] =
              *(const v16h*)&Klds[(jj + kc * 16 + ml) * Ee + ec * 32 + g * 16];
      // B-frags for O += P*V: column = d, K-dim = key (contiguous in Vt row).
      v16h vfr[4];
#pragma unroll
      for (int dc = 0; dc < 4; ++dc)
        vfr[dc] = *(const v16h*)&Vt[(dc * 16 + ml) * TKV + jj + g * 16];

#pragma unroll
      for (int t = 0; t < 2; ++t) {
        v8f zacc = {};
        v8f s0 = WMMA_F16(qa[t][0], kfr[0][0], zacc);
        s0     = WMMA_F16(qa[t][1], kfr[0][1], s0);
        v8f s1 = WMMA_F16(qa[t][0], kfr[1][0], zacc);
        s1     = WMMA_F16(qa[t][1], kfr[1][1], s1);

        // Online softmax over 32 new keys (base-2 domain, raw v_exp_f32).
        float nm[8], alpha[8];
#pragma unroll
        for (int r = 0; r < 8; ++r) {
          float mx = rmax16(fmax_raw(s0[r], s1[r]));
          nm[r]    = fmax_raw(mrow[t][r], mx);
          alpha[r] = __builtin_amdgcn_exp2f(mrow[t][r] - nm[r]);
          mrow[t][r] = nm[r];
        }
#pragma unroll
        for (int r = 0; r < 8; ++r) {
          s0[r] = __builtin_amdgcn_exp2f(s0[r] - nm[r]);
          s1[r] = __builtin_amdgcn_exp2f(s1[r] - nm[r]);
          lrow[t][r] = lrow[t][r] * alpha[r] + rsum16(s0[r] + s1[r]);
        }
#pragma unroll
        for (int dc = 0; dc < 4; ++dc)
#pragma unroll
          for (int r = 0; r < 8; ++r) oacc[t][dc][r] *= alpha[r];

        // D-layout -> A-layout via per-wave LDS staging (same-wave DS is
        // in-order; no workgroup barrier needed).
        _Float16* ps = Pst[wave];
#pragma unroll
        for (int r = 0; r < 8; ++r) {
          ps[(r + 8 * g) * 32 + ml]      = (_Float16)s0[r];
          ps[(r + 8 * g) * 32 + 16 + ml] = (_Float16)s1[r];
        }
        v8h lo = *(const v8h*)&ps[ml * 32 + g * 16];
        v8h hi = *(const v8h*)&ps[ml * 32 + g * 16 + 8];
        v16h pa;
#pragma unroll
        for (int i = 0; i < 8; ++i) { pa[i] = lo[i]; pa[i + 8] = hi[i]; }

#pragma unroll
        for (int dc = 0; dc < 4; ++dc)
          oacc[t][dc] = WMMA_F16(pa, vfr[dc], oacc[t][dc]);
      }
    }
    __syncthreads();  // tile consumed before refill
  }

  // ---- Epilogue: normalize and store (lanes 0..15 write consecutive d).
#pragma unroll
  for (int t = 0; t < 2; ++t)
#pragma unroll
    for (int r = 0; r < 8; ++r) {
      const float inv = 1.0f / lrow[t][r];
      float* orow = ob + (size_t)(wave * 32 + t * 16 + 8 * g + r) * NSTR;
#pragma unroll
      for (int dc = 0; dc < 4; ++dc)
        orow[dc * 16 + ml] = oacc[t][dc][r] * inv;
    }
}

extern "C" void kernel_launch(void* const* d_in, const int* in_sizes, int n_in,
                              void* d_out, int out_size, void* d_ws, size_t ws_size,
                              hipStream_t stream) {
  (void)in_sizes; (void)n_in; (void)out_size; (void)d_ws; (void)ws_size;
  const float* q = (const float*)d_in[0];
  const float* k = (const float*)d_in[1];
  const float* v = (const float*)d_in[2];
  float* o = (float*)d_out;
  sfa_flash_kernel<<<dim3(Bb * Ll * Hh), dim3(256), 0, stream>>>(q, k, v, o);
}
